// SelectiveNeighborhoodAttention_77111842832983
// MI455X (gfx1250) — compile-verified, matrix operand order checked by reading, not run on hardware
//
#include <hip/hip_runtime.h>
#include <hip/hip_bf16.h>

// ---------------------------------------------------------------------------
// SelectiveNeighborhoodAttention for MI455X (gfx1250, wave32, WMMA).
//
//   [0] one-time: x -> bf16; qkv_w/proj_w/g1_w -> transposed bf16 [N][K]
//   [1] qkv   = x @ qkv_w + qkv_b            (WMMA bf16, q *= hd^-0.5)
//   [2] attn  = NATTEN 7x7 softmax(qk+rpb)@v (wave per (token,head)) -> bf16
//   [3] projo = attn @ proj_w + proj_b       (WMMA bf16)
//   [4] g1o   = relu(x @ g1_w + g1_b)        (WMMA bf16, fused ReLU)
//   [5] out   = LN(x + projo * sigmoid(g1o @ g2_w + g2_b)) * ln_g + ln_b
//
// GEMM wave tile: 32(M) x 64(N) = 8 x v_wmma_f32_16x16x32_bf16 per K-step,
// fed by 12 global_load_b128 (A row-major bf16, B pre-transposed bf16 so a
// fragment's K-run is contiguous). Block = 8 waves along M -> 256x64 tile.
// ---------------------------------------------------------------------------

typedef __attribute__((ext_vector_type(16))) __bf16 v16bf;
typedef __attribute__((ext_vector_type(8)))  __bf16 v8bf;
typedef __attribute__((ext_vector_type(4)))  __bf16 v4bf;
typedef __attribute__((ext_vector_type(8)))  float  v8f;

#define TOKENS  8192   // 2*64*64
#define DIMC    256
#define NHEADS  8
#define HDIM    32
#define HH      64
#define WW      64
#define KWIN    7

// round-to-nearest-even f32 -> bf16
__device__ __forceinline__ __bf16 f2bf(float f) {
    unsigned u = __builtin_bit_cast(unsigned, f);
    u += 0x7FFFu + ((u >> 16) & 1u);
    unsigned short s = (unsigned short)(u >> 16);
    return __builtin_bit_cast(__bf16, s);
}

// ---------------------------------------------------------------------------
// [0a] straight f32 -> bf16 (4 elements / thread)
// ---------------------------------------------------------------------------
__global__ void __launch_bounds__(256)
sna_cvt_bf16(const float* __restrict__ src, __bf16* __restrict__ dst, int n4)
{
    int id = blockIdx.x * blockDim.x + threadIdx.x;
    if (id >= n4) return;
    float4 f = ((const float4*)src)[id];
    v4bf o;
    o[0] = f2bf(f.x); o[1] = f2bf(f.y); o[2] = f2bf(f.z); o[3] = f2bf(f.w);
    ((v4bf*)dst)[id] = o;
}

// ---------------------------------------------------------------------------
// [0b] W[K,N] f32  ->  Wt[N,K] bf16  (one thread per output element)
// ---------------------------------------------------------------------------
__global__ void __launch_bounds__(256)
sna_cvt_transpose(const float* __restrict__ W, __bf16* __restrict__ Wt,
                  int K, int N)
{
    int id = blockIdx.x * blockDim.x + threadIdx.x;
    if (id >= K * N) return;
    int n = id / K;
    int k = id - n * K;
    Wt[id] = f2bf(W[k * N + n]);
}

// ---------------------------------------------------------------------------
// WMMA GEMM: C[M,N] = f( A[M,K]bf16 @ B[K,N] + bias )  with B given as
// Bt[N,K] bf16.  Fragment layout per ISA 7.12.2 (16-bit A 16x32): lane L,
// row/col = L&15, element t<8 -> K = 8*(L>>4)+t, t>=8 -> K = 16+8*(L>>4)+t-8.
// Both A and Bt have their fragment K-runs contiguous -> b128 loads.
// ---------------------------------------------------------------------------
__global__ void __launch_bounds__(256)
sna_gemm_bf16_wmma(const __bf16* __restrict__ A, const __bf16* __restrict__ Bt,
                   const float* __restrict__ bias, float* __restrict__ C,
                   int M, int N, int K, float qscale, int scaleCols, int doRelu)
{
    const int lane = threadIdx.x & 31;
    const int wid  = threadIdx.x >> 5;
    const int row0 = blockIdx.y * 256 + wid * 32;   // wave: rows row0..row0+31
    const int col0 = blockIdx.x * 64;               // wave: cols col0..col0+63
    const int mrow = lane & 15;
    const int kk   = (lane >> 4) << 3;              // 0 or 8

    v8f zero = {};
    v8f acc[8];                                      // [mi*4 + ni]
#pragma unroll
    for (int i = 0; i < 8; ++i) acc[i] = zero;

    const __bf16* aBase = A  + (row0 + mrow) * K + kk;
    const __bf16* bBase = Bt + (col0 + mrow) * K + kk;

    for (int k0 = 0; k0 < K; k0 += 32) {
        const __bf16* ap0 = aBase + k0;              // rows +0..15
        const __bf16* ap1 = ap0 + 16 * K;            // rows +16..31

        v16bf a0 = __builtin_shufflevector(*(const v8bf*)ap0,
                                           *(const v8bf*)(ap0 + 16),
                                           0,1,2,3,4,5,6,7,8,9,10,11,12,13,14,15);
        v16bf a1 = __builtin_shufflevector(*(const v8bf*)ap1,
                                           *(const v8bf*)(ap1 + 16),
                                           0,1,2,3,4,5,6,7,8,9,10,11,12,13,14,15);
        v16bf b[4];
#pragma unroll
        for (int ni = 0; ni < 4; ++ni) {
            const __bf16* bp = bBase + ni * 16 * K + k0;
            b[ni] = __builtin_shufflevector(*(const v8bf*)bp,
                                            *(const v8bf*)(bp + 16),
                                            0,1,2,3,4,5,6,7,8,9,10,11,12,13,14,15);
        }

        // global_prefetch_b8 for the next K-tile (gfx1250 prefetch path)
        __builtin_prefetch(ap0 + 32, 0, 0);
        __builtin_prefetch(bBase + k0 + 32, 0, 0);

#pragma unroll
        for (int ni = 0; ni < 4; ++ni) {
            acc[ni]     = __builtin_amdgcn_wmma_f32_16x16x32_bf16(
                              false, a0, false, b[ni], (short)0, acc[ni],
                              false, false);
            acc[4 + ni] = __builtin_amdgcn_wmma_f32_16x16x32_bf16(
                              false, a1, false, b[ni], (short)0, acc[4 + ni],
                              false, false);
        }
    }

    // C/D layout (32-bit 16x16): row = 8*(lane>>4)+t, col = lane&15
    const int cbase = col0 + (lane & 15);
    const int rbase = row0 + ((lane >> 4) << 3);
#pragma unroll
    for (int mi = 0; mi < 2; ++mi) {
#pragma unroll
        for (int ni = 0; ni < 4; ++ni) {
            const v8f a = acc[mi * 4 + ni];
            const int cn = cbase + ni * 16;
            const float bv = bias[cn];
            const float sc = (cn < scaleCols) ? qscale : 1.f;
#pragma unroll
            for (int t = 0; t < 8; ++t) {
                float v = (a[t] + bv) * sc;
                if (doRelu) v = fmaxf(v, 0.f);
                C[(rbase + mi * 16 + t) * N + cn] = v;
            }
        }
    }
}

// ---------------------------------------------------------------------------
// Neighborhood attention: one wave per (token, head); lane == head channel.
// qkv layout: [tok, 768] = q(0..255)|k(256..511)|v(512..767), head-major.
// q already scaled by hd^-0.5 in the GEMM.  Output stored bf16 (feeds WMMA).
// ---------------------------------------------------------------------------
__global__ void __launch_bounds__(256)
sna_natten(const float* __restrict__ qkv, const float* __restrict__ rpb,
           __bf16* __restrict__ attn_out)
{
    const int gwave = (blockIdx.x * blockDim.x + threadIdx.x) >> 5;
    const int lane  = threadIdx.x & 31;
    const int head  = gwave & (NHEADS - 1);
    const int tok   = gwave >> 3;
    const int b  = tok >> 12;
    const int ij = tok & 4095;
    const int i  = ij >> 6;
    const int j  = ij & 63;

    const int si = min(max(i - KWIN / 2, 0), HH - KWIN);
    const int sj = min(max(j - KWIN / 2, 0), WW - KWIN);

    const float qv = qkv[tok * (3 * DIMC) + head * HDIM + lane];

    float sc[KWIN * KWIN];
    float mx = -1e30f;
#pragma unroll
    for (int p = 0; p < KWIN; ++p) {
        const int ki = si + p;
        const int bh = ki - i + (KWIN - 1);
#pragma unroll
        for (int q = 0; q < KWIN; ++q) {
            const int kj = sj + q;
            const int bw = kj - j + (KWIN - 1);
            const int kt = b * (HH * WW) + ki * WW + kj;
            const float kv = qkv[kt * (3 * DIMC) + DIMC + head * HDIM + lane];
            float s = qv * kv;
#pragma unroll
            for (int off = 16; off > 0; off >>= 1)     // wave32 tree reduce
                s += __shfl_xor(s, off, 32);
            s += rpb[head * (2*KWIN-1)*(2*KWIN-1) + bh * (2*KWIN-1) + bw];
            sc[p * KWIN + q] = s;
            mx = fmaxf(mx, s);
        }
    }

    float denom = 0.f;
#pragma unroll
    for (int n = 0; n < KWIN * KWIN; ++n) {
        sc[n] = __expf(sc[n] - mx);
        denom += sc[n];
    }
    const float inv = 1.f / denom;

    float accv = 0.f;
#pragma unroll
    for (int p = 0; p < KWIN; ++p) {
#pragma unroll
        for (int q = 0; q < KWIN; ++q) {
            const int kt = b * (HH * WW) + (si + p) * WW + (sj + q);
            accv += sc[p * KWIN + q] *
                    qkv[kt * (3 * DIMC) + 2 * DIMC + head * HDIM + lane];
        }
    }
    attn_out[tok * DIMC + head * HDIM + lane] = f2bf(accv * inv);
}

// ---------------------------------------------------------------------------
// Epilogue: gate = sigmoid(g1o . g2_w + g2_b); y = x + projo*gate; LayerNorm.
// ---------------------------------------------------------------------------
__global__ void __launch_bounds__(256)
sna_epilogue(const float* __restrict__ x, const float* __restrict__ projo,
             const float* __restrict__ g1o, const float* __restrict__ g2w,
             const float* __restrict__ g2b, const float* __restrict__ lng,
             const float* __restrict__ lnb, float* __restrict__ out)
{
    const int t = blockIdx.x;
    const int c = threadIdx.x;
    __shared__ float s1[DIMC];
    __shared__ float s2[DIMC];
    __shared__ float sgate;

    s1[c] = (c < DIMC / 4) ? g1o[t * (DIMC / 4) + c] * g2w[c] : 0.f;
    __syncthreads();
    for (int s = 128; s > 0; s >>= 1) {
        if (c < s) s1[c] += s1[c + s];
        __syncthreads();
    }
    if (c == 0) sgate = 1.f / (1.f + __expf(-(s1[0] + g2b[0])));
    __syncthreads();
    const float g = sgate;

    const float y = x[t * DIMC + c] + projo[t * DIMC + c] * g;
    s1[c] = y;
    s2[c] = y * y;
    __syncthreads();
    for (int s = 128; s > 0; s >>= 1) {
        if (c < s) { s1[c] += s1[c + s]; s2[c] += s2[c + s]; }
        __syncthreads();
    }
    const float mu  = s1[0] * (1.f / DIMC);
    const float var = s2[0] * (1.f / DIMC) - mu * mu;
    const float r   = rsqrtf(var + 1e-5f);
    out[t * DIMC + c] = (y - mu) * r * lng[c] + lnb[c];
}

// ---------------------------------------------------------------------------
extern "C" void kernel_launch(void* const* d_in, const int* in_sizes, int n_in,
                              void* d_out, int out_size, void* d_ws, size_t ws_size,
                              hipStream_t stream)
{
    const float* x      = (const float*)d_in[0];
    const float* qkv_w  = (const float*)d_in[1];
    const float* qkv_b  = (const float*)d_in[2];
    const float* rpb    = (const float*)d_in[3];
    const float* proj_w = (const float*)d_in[4];
    const float* proj_b = (const float*)d_in[5];
    const float* g1_w   = (const float*)d_in[6];
    const float* g1_b   = (const float*)d_in[7];
    const float* g2_w   = (const float*)d_in[8];
    const float* g2_b   = (const float*)d_in[9];
    const float* ln_g   = (const float*)d_in[10];
    const float* ln_b   = (const float*)d_in[11];
    float*       out    = (float*)d_out;

    // -------- workspace layout --------
    float*  qkv     = (float*)d_ws;                       // 8192*768 f32
    float*  projo   = qkv   + TOKENS * 3 * DIMC;          // 8192*256 f32
    float*  g1o     = projo + TOKENS * DIMC;              // 8192*64  f32
    __bf16* xb      = (__bf16*)(g1o + TOKENS * (DIMC/4)); // 8192*256 bf16
    __bf16* attnb   = xb      + TOKENS * DIMC;            // 8192*256 bf16
    __bf16* wt_qkv  = attnb   + TOKENS * DIMC;            // 768*256  bf16
    __bf16* wt_proj = wt_qkv  + 3 * DIMC * DIMC;          // 256*256  bf16
    __bf16* wt_g1   = wt_proj + DIMC * DIMC;              // 64*256   bf16

    const float qscale = 0.17677669529663687f;            // 32^-0.5
    dim3 blk(256);

    // [0] one-time precision conversion / weight transposition
    sna_cvt_bf16<<<dim3(TOKENS * DIMC / 4 / 256), blk, 0, stream>>>(
        x, xb, TOKENS * DIMC / 4);
    sna_cvt_transpose<<<dim3(DIMC * 3 * DIMC / 256), blk, 0, stream>>>(
        qkv_w, wt_qkv, DIMC, 3 * DIMC);
    sna_cvt_transpose<<<dim3(DIMC * DIMC / 256), blk, 0, stream>>>(
        proj_w, wt_proj, DIMC, DIMC);
    sna_cvt_transpose<<<dim3(DIMC * (DIMC / 4) / 256), blk, 0, stream>>>(
        g1_w, wt_g1, DIMC, DIMC / 4);

    // [1] qkv GEMM (q pre-scaled)
    sna_gemm_bf16_wmma<<<dim3(3 * DIMC / 64, TOKENS / 256), blk, 0, stream>>>(
        xb, wt_qkv, qkv_b, qkv, TOKENS, 3 * DIMC, DIMC, qscale, DIMC, 0);

    // [2] neighborhood attention
    sna_natten<<<dim3(TOKENS), blk, 0, stream>>>(qkv, rpb, attnb);

    // [3] output projection GEMM
    sna_gemm_bf16_wmma<<<dim3(DIMC / 64, TOKENS / 256), blk, 0, stream>>>(
        attnb, wt_proj, proj_b, projo, TOKENS, DIMC, DIMC, 1.f, 0, 0);

    // [4] gate layer-1 GEMM with fused ReLU (N = 64)
    sna_gemm_bf16_wmma<<<dim3(1, TOKENS / 256), blk, 0, stream>>>(
        xb, wt_g1, g1_b, g1o, TOKENS, DIMC / 4, DIMC, 1.f, 0, 1);

    // [5] gate dot + sigmoid + residual + LayerNorm
    sna_epilogue<<<dim3(TOKENS), blk, 0, stream>>>(
        x, projo, g1o, g2_w, g2_b, ln_g, ln_b, out);
}
